// FusedLinearCrossEntropyLoss_13254269075633
// MI455X (gfx1250) — compile-verified, matrix operand order checked by reading, not run on hardware
//
#include <hip/hip_runtime.h>
#include <hip/hip_bf16.h>

// Problem constants (from reference)
constexpr int Nc = 8192;
constexpr int Hc = 1024;
constexpr int Vc = 32000;
constexpr int IGNORE_INDEX = -100;

// Tiling
constexpr int NCHUNK = 25;          // V split across grid.y
constexpr int CV = Vc / NCHUNK;     // 1280 columns per chunk
constexpr int TILES_PER_WAVE = (CV / 16) / 8;  // 80 col-tiles / 8 waves = 10
constexpr int BM = 32;              // rows per block (2 row-tiles per wave)

typedef __attribute__((ext_vector_type(16))) __bf16 v16bf;
typedef __attribute__((ext_vector_type(8)))  __bf16 v8bf;
typedef __attribute__((ext_vector_type(4)))  __bf16 v4bf;
typedef __attribute__((ext_vector_type(8)))  float  v8f;

// ---------------------------------------------------------------------------
// bf16 split helpers.  x = hi + lo with hi = truncated top-16 bits (exact in
// bf16) and lo = bf16(residual); bf16x3 product hi*hi + hi*lo + lo*hi is
// ~fp32-accurate.
// ---------------------------------------------------------------------------
struct bf2 {
  __bf16 h;
  __bf16 l;
};

__device__ __forceinline__ bf2 split1(float f) {
  unsigned u = __builtin_bit_cast(unsigned, f) & 0xFFFF0000u;
  bf2 r;
  r.h = __builtin_bit_cast(__bf16, (unsigned short)(u >> 16));
  r.l = (__bf16)(f - __builtin_bit_cast(float, u));
  return r;
}

// Fragment element order for 16-bit A/B operands (wave32):
//   e in [0,8):  K = k0 + 8*(lane>>4) + e        (VGPR 0..3)
//   e in [8,16): K = k0 + 16 + 8*(lane>>4) + e-8 (VGPR 4..7)
// p must point at row + k0 + 8*(lane>>4); both halves are 16B-aligned.
__device__ __forceinline__ v16bf load_frag_bf16(const __bf16* __restrict__ p) {
  v8bf a = *reinterpret_cast<const v8bf*>(p);
  v8bf b = *reinterpret_cast<const v8bf*>(p + 16);
  return __builtin_shufflevector(a, b, 0, 1, 2, 3, 4, 5, 6, 7,
                                 8, 9, 10, 11, 12, 13, 14, 15);
}

__device__ __forceinline__ void load_split_frag(const float* __restrict__ p,
                                                v16bf& h, v16bf& l) {
#pragma unroll
  for (int e = 0; e < 8; ++e) {
    bf2 r0 = split1(p[e]);
    bf2 r1 = split1(p[e + 16]);
    h[e] = r0.h;
    l[e] = r0.l;
    h[e + 8] = r1.h;
    l[e + 8] = r1.l;
  }
}

__device__ __forceinline__ v8f wmma_bf16(v16bf a, v16bf b, v8f c) {
  return __builtin_amdgcn_wmma_f32_16x16x32_bf16(false, a, false, b, (short)0, c,
                                                 false, false);
}

// ---------------------------------------------------------------------------
// Online softmax update for one 16x16 logits tile held in a C-fragment.
// C layout: vgpr v, lanes 0-15 -> row v, lanes 16-31 -> row v+8; col = lane&15.
// ---------------------------------------------------------------------------
__device__ __forceinline__ void online_update(const v8f& acc, float bval, int colg,
                                              const int* trow8, int rowg_base,
                                              float* m8, float* s8,
                                              float* __restrict__ ws_tgt) {
#pragma unroll
  for (int v = 0; v < 8; ++v) {
    float val = acc[v] + bval;
    if (trow8[v] == colg) ws_tgt[rowg_base + v] = val;  // unique writer grid-wide
    float mt = val;
    mt = fmaxf(mt, __shfl_xor(mt, 1, 32));
    mt = fmaxf(mt, __shfl_xor(mt, 2, 32));
    mt = fmaxf(mt, __shfl_xor(mt, 4, 32));
    mt = fmaxf(mt, __shfl_xor(mt, 8, 32));
    float nm = fmaxf(m8[v], mt);
    float p = __expf(val - nm);
    p += __shfl_xor(p, 1, 32);
    p += __shfl_xor(p, 2, 32);
    p += __shfl_xor(p, 4, 32);
    p += __shfl_xor(p, 8, 32);
    s8[v] = s8[v] * __expf(m8[v] - nm) + p;
    m8[v] = nm;
  }
}

// Shared epilogue: combine 8 waves' (m,s) per row -> partial lse for chunk.
__device__ __forceinline__ void block_combine(int tid, int wv, int lo16, int hi,
                                              int rowblk, int chunk,
                                              float (&m_run)[2][8], float (&s_run)[2][8],
                                              float* __restrict__ ws_lse) {
  __shared__ float lds_m[8][BM];
  __shared__ float lds_s[8][BM];
  if (lo16 == 0) {
#pragma unroll
    for (int rt = 0; rt < 2; ++rt)
#pragma unroll
      for (int v = 0; v < 8; ++v) {
        lds_m[wv][rt * 16 + 8 * hi + v] = m_run[rt][v];
        lds_s[wv][rt * 16 + 8 * hi + v] = s_run[rt][v];
      }
  }
  __syncthreads();
  if (tid < BM) {
    float M = -__builtin_inff();
#pragma unroll
    for (int w = 0; w < 8; ++w) M = fmaxf(M, lds_m[w][tid]);
    float S = 0.0f;
#pragma unroll
    for (int w = 0; w < 8; ++w) S += lds_s[w][tid] * __expf(lds_m[w][tid] - M);
    ws_lse[(size_t)(rowblk + tid) * NCHUNK + chunk] = M + __logf(S);
  }
}

// ---------------------------------------------------------------------------
// Pre-pass: split fp32 array into planar bf16 hi/lo arrays (4 elems/thread).
// ---------------------------------------------------------------------------
__global__ __launch_bounds__(256) void split_bf16_kernel(
    const float* __restrict__ src, __bf16* __restrict__ hi,
    __bf16* __restrict__ lo, int n4) {
  int i = blockIdx.x * blockDim.x + threadIdx.x;
  if (i >= n4) return;
  float4 f = reinterpret_cast<const float4*>(src)[i];
  bf2 r0 = split1(f.x);
  bf2 r1 = split1(f.y);
  bf2 r2 = split1(f.z);
  bf2 r3 = split1(f.w);
  v4bf h, l;
  h[0] = r0.h; l[0] = r0.l;
  h[1] = r1.h; l[1] = r1.l;
  h[2] = r2.h; l[2] = r2.l;
  h[3] = r3.h; l[3] = r3.l;
  reinterpret_cast<v4bf*>(hi)[i] = h;
  reinterpret_cast<v4bf*>(lo)[i] = l;
}

// ---------------------------------------------------------------------------
// Kernel 1 (fast): pre-split bf16 operands -> pure load+WMMA hot loop.
// ---------------------------------------------------------------------------
__global__ __launch_bounds__(256) void flce_logits_fast_kernel(
    const __bf16* __restrict__ xh, const __bf16* __restrict__ xl,
    const __bf16* __restrict__ wh, const __bf16* __restrict__ wl,
    const float* __restrict__ bias, const int* __restrict__ target,
    float* __restrict__ ws_lse, float* __restrict__ ws_tgt) {
  const int tid   = threadIdx.x;
  const int wv    = tid >> 5;
  const int lane  = tid & 31;
  const int lo16  = lane & 15;
  const int hi    = lane >> 4;
  const int kb    = hi * 8;
  const int rowblk   = blockIdx.x * BM;
  const int chunk    = blockIdx.y;
  const int colchunk = chunk * CV;

  const size_t arow = (size_t)(rowblk + lo16) * Hc + kb;
  const __bf16* a0h = xh + arow;
  const __bf16* a0l = xl + arow;
  const __bf16* a1h = a0h + (size_t)16 * Hc;
  const __bf16* a1l = a0l + (size_t)16 * Hc;

  int trow[2][8];
  float m_run[2][8], s_run[2][8];
#pragma unroll
  for (int rt = 0; rt < 2; ++rt)
#pragma unroll
    for (int v = 0; v < 8; ++v) {
      trow[rt][v]  = target[rowblk + rt * 16 + 8 * hi + v];
      m_run[rt][v] = -__builtin_inff();
      s_run[rt][v] = 0.0f;
    }

  for (int i = 0; i < TILES_PER_WAVE; ++i) {
    const int t = wv + (i << 3);                  // col-tile 0..79 within chunk
    const int colbase = colchunk + t * 16;
    const size_t brow = (size_t)(colbase + lo16) * Hc + kb;
    const __bf16* bhp = wh + brow;
    const __bf16* blp = wl + brow;

    v8f acc0 = {0.f, 0.f, 0.f, 0.f, 0.f, 0.f, 0.f, 0.f};
    v8f acc1 = {0.f, 0.f, 0.f, 0.f, 0.f, 0.f, 0.f, 0.f};

#pragma unroll 4
    for (int k0 = 0; k0 < Hc; k0 += 32) {
      v16bf fa0h = load_frag_bf16(a0h + k0);
      v16bf fa0l = load_frag_bf16(a0l + k0);
      v16bf fa1h = load_frag_bf16(a1h + k0);
      v16bf fa1l = load_frag_bf16(a1l + k0);
      v16bf fbh  = load_frag_bf16(bhp + k0);
      v16bf fbl  = load_frag_bf16(blp + k0);
      acc0 = wmma_bf16(fa0h, fbh, acc0);
      acc0 = wmma_bf16(fa0h, fbl, acc0);
      acc0 = wmma_bf16(fa0l, fbh, acc0);
      acc1 = wmma_bf16(fa1h, fbh, acc1);
      acc1 = wmma_bf16(fa1h, fbl, acc1);
      acc1 = wmma_bf16(fa1l, fbh, acc1);
    }

    const int colg   = colbase + lo16;
    const float bval = bias[colg];
    online_update(acc0, bval, colg, trow[0], rowblk + 8 * hi,      m_run[0], s_run[0], ws_tgt);
    online_update(acc1, bval, colg, trow[1], rowblk + 16 + 8 * hi, m_run[1], s_run[1], ws_tgt);
  }

  block_combine(tid, wv, lo16, hi, rowblk, chunk, m_run, s_run, ws_lse);
}

// ---------------------------------------------------------------------------
// Kernel 1 (fallback, small ws): convert fp32->bf16 hi/lo in the loop.
// ---------------------------------------------------------------------------
__global__ __launch_bounds__(256) void flce_logits_direct_kernel(
    const float* __restrict__ x, const float* __restrict__ wt,
    const float* __restrict__ bias, const int* __restrict__ target,
    float* __restrict__ ws_lse, float* __restrict__ ws_tgt) {
  const int tid   = threadIdx.x;
  const int wv    = tid >> 5;
  const int lane  = tid & 31;
  const int lo16  = lane & 15;
  const int hi    = lane >> 4;
  const int kb    = hi * 8;
  const int rowblk   = blockIdx.x * BM;
  const int chunk    = blockIdx.y;
  const int colchunk = chunk * CV;

  const float* arow0 = x + (size_t)(rowblk + lo16) * Hc;
  const float* arow1 = arow0 + (size_t)16 * Hc;

  int trow[2][8];
  float m_run[2][8], s_run[2][8];
#pragma unroll
  for (int rt = 0; rt < 2; ++rt)
#pragma unroll
    for (int v = 0; v < 8; ++v) {
      trow[rt][v]  = target[rowblk + rt * 16 + 8 * hi + v];
      m_run[rt][v] = -__builtin_inff();
      s_run[rt][v] = 0.0f;
    }

  for (int i = 0; i < TILES_PER_WAVE; ++i) {
    const int t = wv + (i << 3);
    const int colbase = colchunk + t * 16;
    const float* brow = wt + (size_t)(colbase + lo16) * Hc;

    v8f acc0 = {0.f, 0.f, 0.f, 0.f, 0.f, 0.f, 0.f, 0.f};
    v8f acc1 = {0.f, 0.f, 0.f, 0.f, 0.f, 0.f, 0.f, 0.f};

#pragma unroll 2
    for (int k0 = 0; k0 < Hc; k0 += 32) {
      v16bf fa0h, fa0l, fa1h, fa1l, fbh, fbl;
      load_split_frag(arow0 + k0 + kb, fa0h, fa0l);
      load_split_frag(arow1 + k0 + kb, fa1h, fa1l);
      load_split_frag(brow + k0 + kb, fbh, fbl);
      acc0 = wmma_bf16(fa0h, fbh, acc0);
      acc0 = wmma_bf16(fa0h, fbl, acc0);
      acc0 = wmma_bf16(fa0l, fbh, acc0);
      acc1 = wmma_bf16(fa1h, fbh, acc1);
      acc1 = wmma_bf16(fa1h, fbl, acc1);
      acc1 = wmma_bf16(fa1l, fbh, acc1);
    }

    const int colg   = colbase + lo16;
    const float bval = bias[colg];
    online_update(acc0, bval, colg, trow[0], rowblk + 8 * hi,      m_run[0], s_run[0], ws_tgt);
    online_update(acc1, bval, colg, trow[1], rowblk + 16 + 8 * hi, m_run[1], s_run[1], ws_tgt);
  }

  block_combine(tid, wv, lo16, hi, rowblk, chunk, m_run, s_run, ws_lse);
}

// Kernel 2: merge chunk-partial lses per row, form per-row loss.
__global__ __launch_bounds__(256) void flce_lse_kernel(
    const int* __restrict__ target, const float* __restrict__ ws_lse,
    const float* __restrict__ ws_tgt, float* __restrict__ ws_loss) {
  int r = blockIdx.x * blockDim.x + threadIdx.x;
  if (r >= Nc) return;
  const float* p = ws_lse + (size_t)r * NCHUNK;
  float M = -__builtin_inff();
#pragma unroll
  for (int c = 0; c < NCHUNK; ++c) M = fmaxf(M, p[c]);
  float S = 0.0f;
#pragma unroll
  for (int c = 0; c < NCHUNK; ++c) S += __expf(p[c] - M);
  float lse = M + __logf(S);
  int t = target[r];
  ws_loss[r] = (t != IGNORE_INDEX) ? (lse - ws_tgt[r]) : 0.0f;
}

// Kernel 3: deterministic mean reduction -> out[0]
__global__ __launch_bounds__(256) void flce_reduce_kernel(
    const int* __restrict__ target, const float* __restrict__ ws_loss,
    float* __restrict__ out) {
  __shared__ float ssum[256];
  __shared__ int   scnt[256];
  const int tid = threadIdx.x;
  float s = 0.0f;
  int   c = 0;
  for (int i = tid; i < Nc; i += 256) {
    s += ws_loss[i];
    c += (target[i] != IGNORE_INDEX) ? 1 : 0;
  }
  ssum[tid] = s;
  scnt[tid] = c;
  __syncthreads();
  for (int off = 128; off > 0; off >>= 1) {
    if (tid < off) {
      ssum[tid] += ssum[tid + off];
      scnt[tid] += scnt[tid + off];
    }
    __syncthreads();
  }
  if (tid == 0) {
    int cnt = scnt[0] > 1 ? scnt[0] : 1;
    out[0] = ssum[0] / (float)cnt;
  }
}

extern "C" void kernel_launch(void* const* d_in, const int* in_sizes, int n_in,
                              void* d_out, int out_size, void* d_ws, size_t ws_size,
                              hipStream_t stream) {
  (void)in_sizes; (void)n_in; (void)out_size;
  const float* x    = (const float*)d_in[0];   // [N,H]
  const float* w    = (const float*)d_in[1];   // [V,H]
  const float* bias = (const float*)d_in[2];   // [V]
  const int*   tgt  = (const int*)d_in[3];     // [N]
  float* out = (float*)d_out;

  // Workspace layout (small arrays first, 16B-aligned throughout)
  float* ws_lse  = (float*)d_ws;                      // N*NCHUNK floats
  float* ws_tgt  = ws_lse + (size_t)Nc * NCHUNK;      // N floats
  float* ws_loss = ws_tgt + Nc;                       // N floats
  const size_t small_bytes = ((size_t)Nc * NCHUNK + 2 * (size_t)Nc) * sizeof(float);

  __bf16* xh = (__bf16*)((char*)d_ws + small_bytes);  // N*H
  __bf16* xl = xh + (size_t)Nc * Hc;                  // N*H
  __bf16* wh = xl + (size_t)Nc * Hc;                  // V*H
  __bf16* wl = wh + (size_t)Vc * Hc;                  // V*H
  const size_t need_fast = small_bytes +
      2 * (size_t)Nc * Hc * sizeof(__bf16) + 2 * (size_t)Vc * Hc * sizeof(__bf16);

  dim3 g1(Nc / BM, NCHUNK);
  if (ws_size >= need_fast) {
    // One-shot fp32 -> planar bf16 hi/lo split (memory-bound)
    int xn4 = (Nc * Hc) / 4;
    int wn4 = (Vc * Hc) / 4;
    split_bf16_kernel<<<(xn4 + 255) / 256, 256, 0, stream>>>(x, xh, xl, xn4);
    split_bf16_kernel<<<(wn4 + 255) / 256, 256, 0, stream>>>(w, wh, wl, wn4);
    flce_logits_fast_kernel<<<g1, 256, 0, stream>>>(xh, xl, wh, wl, bias, tgt,
                                                    ws_lse, ws_tgt);
  } else {
    flce_logits_direct_kernel<<<g1, 256, 0, stream>>>(x, w, bias, tgt,
                                                      ws_lse, ws_tgt);
  }
  flce_lse_kernel<<<(Nc + 255) / 256, 256, 0, stream>>>(tgt, ws_lse, ws_tgt, ws_loss);
  flce_reduce_kernel<<<1, 256, 0, stream>>>(tgt, ws_loss, out);
}
